// HeterogeneousGraphTransformer_90220083019927
// MI455X (gfx1250) — compile-verified, matrix operand order checked by reading, not run on hardware
//
#include <hip/hip_runtime.h>
#include <math.h>

// ---------------------------------------------------------------------------
// CDNA5 (gfx1250, wave32) WMMA types
// ---------------------------------------------------------------------------
typedef __attribute__((ext_vector_type(16))) _Float16 v16h;
typedef __attribute__((ext_vector_type(8)))  float    v8f;
typedef int v4i __attribute__((vector_size(16)));   // matches builtin param type

#define MT 128          // macro tile M
#define NT 64           // macro tile N
#define KT 32           // K step (WMMA depth)
#define LDH 40          // halves per LDS row (80B: 16B-aligned, 20-dword stride)

// Async global->LDS path (gfx1250): guarded so builtin absence degrades to a
// plain ds-store staging path without breaking compile.
#if __has_builtin(__builtin_amdgcn_global_load_async_to_lds_b128) && \
    __has_builtin(__builtin_amdgcn_s_wait_asynccnt)
#define USE_ASYNC 1
#define COPY16(l, g)                                                        \
    __builtin_amdgcn_global_load_async_to_lds_b128(                         \
        (__attribute__((address_space(1))) v4i*)(g),                        \
        (__attribute__((address_space(3))) v4i*)(l), 0, 0)
#define WAIT_ASYNC() __builtin_amdgcn_s_wait_asynccnt(0)
#else
#define USE_ASYNC 0
#define COPY16(l, g) (*(uint4*)(l) = *(const uint4*)(g))
#define WAIT_ASYNC() ((void)0)
#endif

__device__ __forceinline__ void stage_tile(const _Float16* __restrict__ A,
                                           const _Float16* __restrict__ Bt,
                                           _Float16* sA, _Float16* sB,
                                           int tid, int m0, int n0,
                                           int M, int K, int k0)
{
    for (int idx = tid; idx < 768; idx += 256) {
        if (idx < 512) {
            int row = idx >> 2, c = idx & 3;
            int gm = m0 + row;
            if (gm < M)
                COPY16(sA + row * LDH + c * 8, A + (size_t)gm * K + k0 + c * 8);
        } else {
            int j = idx - 512;
            int n = j >> 2, c = j & 3;
            COPY16(sB + n * LDH + c * 8, Bt + (size_t)(n0 + n) * K + k0 + c * 8);
        }
    }
}

// ---------------------------------------------------------------------------
// GEMM: C[M,N] = act(A[M,K] @ W[K,N] + bias)
// A: f16 row-major; Bt: f16 [N,K] (W pre-transposed); C,bias: f32.
// 8 waves (4x2), wave -> 32x32 outputs, double-buffered LDS, async staging
// of tile t+1 overlapped with WMMA on tile t.
// ---------------------------------------------------------------------------
template<int ACT>  // 0 = none, 1 = relu
__global__ __launch_bounds__(256)
void gemm_f16_wmma(const _Float16* __restrict__ A, const _Float16* __restrict__ Bt,
                   const float* __restrict__ bias, float* __restrict__ C,
                   int M, int N, int K)
{
    __shared__ _Float16 sA[2][MT * LDH];
    __shared__ _Float16 sB[2][NT * LDH];

    const int tid   = threadIdx.x;
    const int lane  = tid & 31;
    const int wave  = tid >> 5;
    const int wm    = wave & 3;
    const int wn    = wave >> 2;
    const int m0    = blockIdx.x * MT;
    const int n0    = blockIdx.y * NT;
    const int row16 = lane & 15;
    const int khalf = lane >> 4;

    v8f acc[2][2] = {};

    stage_tile(A, Bt, sA[0], sB[0], tid, m0, n0, M, K, 0);

    const int nk = K / KT;
    for (int t = 0; t < nk; ++t) {
        const int cur = t & 1;
        WAIT_ASYNC();            // my async copies for tile t have landed
        __syncthreads();         // everyone's landed; tile t-1 compute done
        if (t + 1 < nk)
            stage_tile(A, Bt, sA[cur ^ 1], sB[cur ^ 1], tid, m0, n0, M, K,
                       (t + 1) * KT);

        union FragU { v16h h; unsigned u[8]; };
        FragU fa[2], fb[2];
        #pragma unroll
        for (int mi = 0; mi < 2; ++mi) {
            const _Float16* base = &sA[cur][(wm * 32 + mi * 16 + row16) * LDH];
            #pragma unroll
            for (int j = 0; j < 8; ++j) {
                int kk = ((j < 4) ? 0 : 16) + khalf * 8 + 2 * (j & 3);
                fa[mi].u[j] = *(const unsigned*)(base + kk);
            }
        }
        #pragma unroll
        for (int ni = 0; ni < 2; ++ni) {
            const _Float16* base = &sB[cur][(wn * 32 + ni * 16 + row16) * LDH];
            #pragma unroll
            for (int j = 0; j < 8; ++j) {
                int kk = ((j < 4) ? 0 : 16) + khalf * 8 + 2 * (j & 3);
                fb[ni].u[j] = *(const unsigned*)(base + kk);
            }
        }

        #pragma unroll
        for (int mi = 0; mi < 2; ++mi)
            #pragma unroll
            for (int ni = 0; ni < 2; ++ni)
                acc[mi][ni] = __builtin_amdgcn_wmma_f32_16x16x32_f16(
                    false, fa[mi].h, false, fb[ni].h,
                    (short)0, acc[mi][ni], false, false);

        __syncthreads();
    }

    const bool full = (m0 + MT) <= M;
    #pragma unroll
    for (int mi = 0; mi < 2; ++mi) {
        #pragma unroll
        for (int ni = 0; ni < 2; ++ni) {
            int col = n0 + wn * 32 + ni * 16 + row16;
            float b = bias ? bias[col] : 0.f;
            int rbase = m0 + wm * 32 + mi * 16 + 8 * khalf;
            if (full) {
                #pragma unroll
                for (int r = 0; r < 8; ++r) {
                    float v = acc[mi][ni][r] + b;
                    if (ACT == 1) v = fmaxf(v, 0.f);
                    C[(size_t)(rbase + r) * N + col] = v;
                }
            } else {
                #pragma unroll
                for (int r = 0; r < 8; ++r) {
                    if (rbase + r < M) {
                        float v = acc[mi][ni][r] + b;
                        if (ACT == 1) v = fmaxf(v, 0.f);
                        C[(size_t)(rbase + r) * N + col] = v;
                    }
                }
            }
        }
    }
}

// ---------------------------------------------------------------------------
// Conversion / elementwise kernels
// ---------------------------------------------------------------------------
__global__ void cvt_f32_f16(const float* __restrict__ s, _Float16* __restrict__ d, size_t n) {
    size_t i = (size_t)blockIdx.x * blockDim.x + threadIdx.x;
    if (i < n) d[i] = (_Float16)s[i];
}
__global__ void transpose_cvt(const float* __restrict__ W, _Float16* __restrict__ Wt,
                              int K, int N) {
    size_t i = (size_t)blockIdx.x * blockDim.x + threadIdx.x;
    if (i >= (size_t)K * N) return;
    int k = (int)(i / N), n = (int)(i % N);
    Wt[(size_t)n * K + k] = (_Float16)W[i];
}
__global__ void fill_f32(float* p, size_t n, float v) {
    size_t i = (size_t)blockIdx.x * blockDim.x + threadIdx.x;
    if (i < n) p[i] = v;
}
__global__ void fill_u32(unsigned* p, size_t n, unsigned v) {
    size_t i = (size_t)blockIdx.x * blockDim.x + threadIdx.x;
    if (i < n) p[i] = v;
}
__global__ void gelu_cvt(const float* __restrict__ s, _Float16* __restrict__ d, size_t n) {
    size_t i = (size_t)blockIdx.x * blockDim.x + threadIdx.x;
    if (i >= n) return;
    float x = s[i];
    float t = 0.7978845608028654f * (x + 0.044715f * x * x * x);
    d[i] = (_Float16)(0.5f * x * (1.f + tanhf(t)));
}
__global__ void skip_combine_relu(_Float16* x, const float* o, const float* skip, size_t n) {
    size_t i = (size_t)blockIdx.x * blockDim.x + threadIdx.x;
    if (i >= n) return;
    float alpha = 1.f / (1.f + expf(-skip[0]));
    float v = alpha * o[i] + (1.f - alpha) * (float)x[i];
    x[i] = (_Float16)fmaxf(v, 0.f);
}

__device__ __forceinline__ unsigned enc_f(float f) {
    unsigned u = __float_as_uint(f);
    return (u & 0x80000000u) ? ~u : (u | 0x80000000u);
}
__device__ __forceinline__ float dec_f(unsigned u) {
    u = (u & 0x80000000u) ? (u & 0x7FFFFFFFu) : ~u;
    return __uint_as_float(u);
}
#define ENC_NEG_INF 0x007FFFFFu

#define HID 256
#define HEADS 8
#define DH 32

__global__ void att_pass1(const _Float16* __restrict__ kbuf, const _Float16* __restrict__ qbuf,
                          const int* __restrict__ ei, int E,
                          const float* __restrict__ a_rel, const float* __restrict__ p_rel,
                          float* __restrict__ att, unsigned* __restrict__ menc)
{
    int i = blockIdx.x * blockDim.x + threadIdx.x;
    if (i >= E * HEADS) return;
    int e = i >> 3, h = i & 7;
    int s = ei[e], d = ei[E + e];
    const _Float16* kv = kbuf + (size_t)s * HID + h * DH;
    const _Float16* qv = qbuf + (size_t)d * HID + h * DH;
    const float* a = a_rel + h * DH * DH;
    float q[DH];
    for (int f = 0; f < DH; ++f) q[f] = (float)qv[f];
    float acc = 0.f;
    for (int dd = 0; dd < DH; ++dd) {
        const float* arow = a + dd * DH;
        float t = 0.f;
        for (int f = 0; f < DH; ++f) t += arow[f] * q[f];
        acc += (float)kv[dd] * t;
    }
    float attv = acc * p_rel[h] * 0.17677669529663687f;
    att[i] = attv;
    atomicMax(menc + (size_t)d * HEADS + h, enc_f(attv));
}

__global__ void att_pass2(float* __restrict__ att, const int* __restrict__ ei, int E,
                          const unsigned* __restrict__ menc, float* __restrict__ sbuf)
{
    int i = blockIdx.x * blockDim.x + threadIdx.x;
    if (i >= E * HEADS) return;
    int e = i >> 3, h = i & 7;
    int d = ei[E + e];
    float m = dec_f(menc[(size_t)d * HEADS + h]);
    float ev = expf(att[i] - m);
    att[i] = ev;
    atomicAdd(sbuf + (size_t)d * HEADS + h, ev);
}

__global__ void att_pass3(const float* __restrict__ att, const int* __restrict__ ei, int E,
                          const float* __restrict__ sbuf, const _Float16* __restrict__ vbuf,
                          const float* __restrict__ m_rel, float* __restrict__ outbuf)
{
    int i = blockIdx.x * blockDim.x + threadIdx.x;
    if (i >= E * HEADS) return;
    int e = i >> 3, h = i & 7;
    int s = ei[e], d = ei[E + e];
    float w = att[i] / (sbuf[(size_t)d * HEADS + h] + 1e-16f);
    const _Float16* vv = vbuf + (size_t)s * HID + h * DH;
    const float* mm = m_rel + h * DH * DH;
    float* o = outbuf + (size_t)d * HID + h * DH;
    float vf[DH];
    for (int dd = 0; dd < DH; ++dd) vf[dd] = (float)vv[dd];
    for (int f = 0; f < DH; ++f) {
        float t = 0.f;
        for (int dd = 0; dd < DH; ++dd) t += vf[dd] * mm[dd * DH + f];
        atomicAdd(o + f, t * w);
    }
}

__global__ void gather_pairs(const _Float16* __restrict__ xh, const _Float16* __restrict__ xs,
                             const int* __restrict__ p1, const int* __restrict__ p2,
                             _Float16* __restrict__ pe, int P)
{
    int i = blockIdx.x * blockDim.x + threadIdx.x;
    if (i >= P * 2 * HID) return;
    int p = i >> 9, c = i & 511;
    pe[i] = (c < HID) ? xh[(size_t)p1[p] * HID + c]
                      : xs[(size_t)p2[p] * HID + (c - HID)];
}

__global__ void head_final(const float* __restrict__ hbuf, const float* __restrict__ W2,
                           const float* __restrict__ b2, float* __restrict__ out, int P)
{
    int p = blockIdx.x * blockDim.x + threadIdx.x;
    if (p >= P) return;
    const float* hr = hbuf + (size_t)p * HID;
    float acc = b2[0];
    for (int c = 0; c < HID; ++c) acc += hr[c] * W2[c];
    out[p] = 1.f / (1.f + expf(-acc));
}

// ---------------------------------------------------------------------------
// Host-side orchestration
// ---------------------------------------------------------------------------
static inline unsigned gsz(size_t n) { return (unsigned)((n + 255) / 256); }

static inline void run_gemm(const _Float16* A, const float* W, const float* b, float* C,
                            int M, int N, int K, int act, _Float16* wt, hipStream_t s)
{
    transpose_cvt<<<gsz((size_t)K * N), 256, 0, s>>>(W, wt, K, N);
    dim3 g((M + MT - 1) / MT, N / NT), blk(256);
    if (act) gemm_f16_wmma<1><<<g, blk, 0, s>>>(A, wt, b, C, M, N, K);
    else     gemm_f16_wmma<0><<<g, blk, 0, s>>>(A, wt, b, C, M, N, K);
}

extern "C" void kernel_launch(void* const* d_in, const int* in_sizes, int n_in,
                              void* d_out, int out_size, void* d_ws, size_t ws_size,
                              hipStream_t stream)
{
    (void)n_in; (void)ws_size; (void)out_size;
    auto F = [&](int i) { return (const float*)d_in[i]; };

    const float* x_host_in = F(0);
    const float* x_srv_in  = F(1);
    const int* ei_connects = (const int*)d_in[2];
    const int* ei_runs     = (const int*)d_in[3];
    const int* ei_depends  = (const int*)d_in[4];
    const int* ei_hosted   = (const int*)d_in[5];
    const int* pair1       = (const int*)d_in[6];
    const int* pair2       = (const int*)d_in[7];

    const int NH = in_sizes[0] / 128;
    const int NS = in_sizes[1] / 128;
    const int P  = in_sizes[6];
    const int Ec = in_sizes[2] / 2, Er = in_sizes[3] / 2;
    const int Ed = in_sizes[4] / 2, Eh = in_sizes[5] / 2;
    const int Nn[2] = { NH, NS };
    const int Emax  = max(max(Ec, Er), max(Ed, Eh));
    const int Nmax  = max(NH, NS);

    const float* encW[2] = { F(8),  F(10) };
    const float* encB[2] = { F(9),  F(11) };
    const float* headW1 = F(12); const float* headB1 = F(13);
    const float* headW2 = F(14); const float* headB2 = F(15);

    char* wsb = (char*)d_ws;
    size_t off = 0;
    auto alloc = [&](size_t bytes) {
        void* p = wsb + off;
        off = (off + bytes + 255) & ~(size_t)255;
        return p;
    };
    _Float16* xf[2]; xf[0] = (_Float16*)alloc((size_t)NH * HID * 2);
                     xf[1] = (_Float16*)alloc((size_t)NS * HID * 2);
    _Float16 *kb[2], *qb[2], *vb[2];
    for (int nt = 0; nt < 2; ++nt) {
        kb[nt] = (_Float16*)alloc((size_t)Nn[nt] * HID * 2);
        qb[nt] = (_Float16*)alloc((size_t)Nn[nt] * HID * 2);
        vb[nt] = (_Float16*)alloc((size_t)Nn[nt] * HID * 2);
    }
    float* ob[2]; ob[0] = (float*)alloc((size_t)NH * HID * 4);
                  ob[1] = (float*)alloc((size_t)NS * HID * 4);
    _Float16* gb  = (_Float16*)alloc((size_t)Nmax * HID * 2);
    float*    tmp = (float*)alloc((size_t)Nmax * HID * 4);
    size_t a16n = (size_t)Nmax * 128;
    if ((size_t)P * 2 * HID > a16n) a16n = (size_t)P * 2 * HID;
    _Float16* a16 = (_Float16*)alloc(a16n * 2);
    _Float16* wt  = (_Float16*)alloc((size_t)(2 * HID) * HID * 2);
    float*    att = (float*)alloc((size_t)Emax * HEADS * 4);
    unsigned* menc = (unsigned*)alloc((size_t)Nmax * HEADS * 4);
    float*    sbuf = (float*)alloc((size_t)Nmax * HEADS * 4);
    float*    hbuf = (float*)alloc((size_t)P * HID * 4);

    // ---- encoder: cvt input -> f16, GEMM -> f32 tmp, cvt -> f16 xf --------
    cvt_f32_f16<<<gsz((size_t)NH * 128), 256, 0, stream>>>(x_host_in, a16, (size_t)NH * 128);
    run_gemm(a16, encW[0], encB[0], tmp, NH, HID, 128, 0, wt, stream);
    cvt_f32_f16<<<gsz((size_t)NH * HID), 256, 0, stream>>>(tmp, xf[0], (size_t)NH * HID);
    cvt_f32_f16<<<gsz((size_t)NS * 128), 256, 0, stream>>>(x_srv_in, a16, (size_t)NS * 128);
    run_gemm(a16, encW[1], encB[1], tmp, NS, HID, 128, 0, wt, stream);
    cvt_f32_f16<<<gsz((size_t)NS * HID), 256, 0, stream>>>(tmp, xf[1], (size_t)NS * HID);

    struct Rel { const int* ei; int E; int src; int dst; int off; };
    Rel rels[4] = {
        { ei_connects, Ec, 0, 0, 12 },  // connects : host -> host
        { ei_runs,     Er, 0, 1, 21 },  // runs     : host -> service
        { ei_depends,  Ed, 1, 1, 15 },  // depends  : service -> service
        { ei_hosted,   Eh, 1, 0, 18 },  // hosted_by: service -> host
    };

    for (int L = 0; L < 3; ++L) {
        int base = 16 + L * 30;
        const float* aW[2] = { F(base + 0),  F(base + 2)  };
        const float* aB[2] = { F(base + 1),  F(base + 3)  };
        const float* kW[2] = { F(base + 4),  F(base + 6)  };
        const float* kB[2] = { F(base + 5),  F(base + 7)  };
        const float* qW[2] = { F(base + 8),  F(base + 10) };
        const float* qB[2] = { F(base + 9),  F(base + 11) };
        const float* sk[2] = { F(base + 24), F(base + 25) };
        const float* vW[2] = { F(base + 26), F(base + 28) };
        const float* vB[2] = { F(base + 27), F(base + 29) };

        for (int nt = 0; nt < 2; ++nt) {
            int M = Nn[nt];
            size_t n = (size_t)M * HID;
            run_gemm(xf[nt], kW[nt], kB[nt], tmp, M, HID, HID, 0, wt, stream);
            cvt_f32_f16<<<gsz(n), 256, 0, stream>>>(tmp, kb[nt], n);
            run_gemm(xf[nt], qW[nt], qB[nt], tmp, M, HID, HID, 0, wt, stream);
            cvt_f32_f16<<<gsz(n), 256, 0, stream>>>(tmp, qb[nt], n);
            run_gemm(xf[nt], vW[nt], vB[nt], tmp, M, HID, HID, 0, wt, stream);
            cvt_f32_f16<<<gsz(n), 256, 0, stream>>>(tmp, vb[nt], n);
            fill_f32<<<gsz(n), 256, 0, stream>>>(ob[nt], n, 0.f);
        }

        for (int r = 0; r < 4; ++r) {
            const Rel& R = rels[r];
            const float* a_rel = F(base + R.off);
            const float* m_rel = F(base + R.off + 1);
            const float* p_rel = F(base + R.off + 2);
            size_t segn = (size_t)Nn[R.dst] * HEADS;
            fill_u32<<<gsz(segn), 256, 0, stream>>>(menc, segn, ENC_NEG_INF);
            fill_f32<<<gsz(segn), 256, 0, stream>>>(sbuf, segn, 0.f);
            unsigned g = gsz((size_t)R.E * HEADS);
            att_pass1<<<g, 256, 0, stream>>>(kb[R.src], qb[R.dst], R.ei, R.E,
                                             a_rel, p_rel, att, menc);
            att_pass2<<<g, 256, 0, stream>>>(att, R.ei, R.E, menc, sbuf);
            att_pass3<<<g, 256, 0, stream>>>(att, R.ei, R.E, sbuf, vb[R.src],
                                             m_rel, ob[R.dst]);
        }

        for (int nt = 0; nt < 2; ++nt) {
            size_t n = (size_t)Nn[nt] * HID;
            gelu_cvt<<<gsz(n), 256, 0, stream>>>(ob[nt], gb, n);
            run_gemm(gb, aW[nt], aB[nt], tmp, Nn[nt], HID, HID, 0, wt, stream);
            skip_combine_relu<<<gsz(n), 256, 0, stream>>>(xf[nt], tmp, sk[nt], n);
        }
    }

    // ---- head -------------------------------------------------------------
    _Float16* pe = a16;
    gather_pairs<<<gsz((size_t)P * 2 * HID), 256, 0, stream>>>(xf[0], xf[1],
                                                               pair1, pair2, pe, P);
    run_gemm(pe, headW1, headB1, hbuf, P, HID, 2 * HID, 1, wt, stream);
    head_final<<<gsz(P), 256, 0, stream>>>(hbuf, headW2, headB2, (float*)d_out, P);
}
